// PrototypicalHead_76699525972430
// MI455X (gfx1250) — matrix operand order repaired and checked by reading, not compiled, not run on hardware
//
#include <hip/hip_runtime.h>
#include <hip/hip_bf16.h>

typedef __attribute__((ext_vector_type(16))) _Float16 v16h;
typedef __attribute__((ext_vector_type(8)))  _Float16 v8h;
typedef __attribute__((ext_vector_type(4)))  _Float16 v4h;
typedef __attribute__((ext_vector_type(8)))  float    v8f;

#define D_DIM 1024
#define C_CLS 256
#define EPSN 1e-8f
#define CASTRIDE 257   // pad so M and M+8 rows hit different LDS banks

__device__ __forceinline__ float waveSum32(float v) {
#pragma unroll
  for (int m = 16; m >= 1; m >>= 1) v += __shfl_xor(v, m, 32);
  return v;
}
__device__ __forceinline__ float waveMax32(float v) {
#pragma unroll
  for (int m = 16; m >= 1; m >>= 1) v = fmaxf(v, __shfl_xor(v, m, 32));
  return v;
}
// 256-thread block sum (8 waves)
__device__ __forceinline__ float blockSum256(float v, float* red, int tid) {
  v = waveSum32(v);
  if ((tid & 31) == 0) red[tid >> 5] = v;
  __syncthreads();
  float r = red[0] + red[1] + red[2] + red[3] + red[4] + red[5] + red[6] + red[7];
  __syncthreads();
  return r;
}
__device__ __forceinline__ v8f vzero8() {
  v8f z;
#pragma unroll
  for (int i = 0; i < 8; ++i) z[i] = 0.0f;
  return z;
}

// ---------------- small utility kernels ----------------
__global__ void kZero(float* __restrict__ p, int n) {
  int i = blockIdx.x * blockDim.x + threadIdx.x;
  int stride = gridDim.x * blockDim.x;
  for (; i < n; i += stride) p[i] = 0.0f;
}

__global__ void kHist(const int* __restrict__ lab, float* __restrict__ counts, int S) {
  int i = blockIdx.x * blockDim.x + threadIdx.x;
  if (i < S) atomicAdd(&counts[lab[i]], 1.0f);
}

// Accumulate column sums of L2-normalized rows: mu_sum[d] += x[r][d]/||x[r]||
// One block handles 64 consecutive rows; per-thread register column partials,
// then a single global atomic per column per block.
__global__ void __launch_bounds__(256) kMu(const float* __restrict__ X,
                                           float* __restrict__ mu_sum) {
  __shared__ float red[8];
  int tid = threadIdx.x;
  int c = tid * 4;
  int r0 = blockIdx.x * 64;
  float a0 = 0.f, a1 = 0.f, a2 = 0.f, a3 = 0.f;
  for (int r = r0; r < r0 + 64; ++r) {
    const float4 x = *(const float4*)(X + (size_t)r * D_DIM + c);
    float ss = blockSum256(x.x * x.x + x.y * x.y + x.z * x.z + x.w * x.w, red, tid);
    float inv = 1.0f / fmaxf(sqrtf(ss), EPSN);
    a0 += x.x * inv; a1 += x.y * inv; a2 += x.z * inv; a3 += x.w * inv;
  }
  atomicAdd(&mu_sum[c + 0], a0);
  atomicAdd(&mu_sum[c + 1], a1);
  atomicAdd(&mu_sum[c + 2], a2);
  atomicAdd(&mu_sum[c + 3], a3);
}

// y = normalize(normalize(x) - mu), written as f16. One block per row.
__global__ void __launch_bounds__(256) kPack(const float* __restrict__ X,
                                             _Float16* __restrict__ Y,
                                             const float* __restrict__ mu_sum,
                                             float invS) {
  __shared__ float red[8];
  int tid = threadIdx.x;
  int c = tid * 4;
  size_t r = blockIdx.x;
  const float4 x = *(const float4*)(X + r * D_DIM + c);
  float ss = blockSum256(x.x * x.x + x.y * x.y + x.z * x.z + x.w * x.w, red, tid);
  float inv1 = 1.0f / fmaxf(sqrtf(ss), EPSN);
  float y0 = x.x * inv1 - mu_sum[c + 0] * invS;
  float y1 = x.y * inv1 - mu_sum[c + 1] * invS;
  float y2 = x.z * inv1 - mu_sum[c + 2] * invS;
  float y3 = x.w * inv1 - mu_sum[c + 3] * invS;
  float ss2 = blockSum256(y0 * y0 + y1 * y1 + y2 * y2 + y3 * y3, red, tid);
  float inv2 = 1.0f / fmaxf(sqrtf(ss2), EPSN);
  v4h o;
  o[0] = (_Float16)(y0 * inv2);
  o[1] = (_Float16)(y1 * inv2);
  o[2] = (_Float16)(y2 * inv2);
  o[3] = (_Float16)(y3 * inv2);
  *(v4h*)(Y + r * D_DIM + c) = o;
}

// ---------------- fused WMMA GEMM + exp + per-class scatter ----------------
// Block = 256 threads (8 waves). Each block owns 32 query rows and one half of
// the support set (S-split of 2). Each wave walks 32-support "pairs" (two 16x16
// N-tiles), accumulating 4 independent f32 WMMA tiles (2 q-tiles x 2 s-tiles)
// over K=1024 in steps of 32 with v_wmma_f32_16x16x32_f16. exp(sims*scale) is
// scattered into a padded LDS class-mass table with ds_add_f32 atomics, then
// flushed (no global atomics) to a per-split partial buffer CM[2][Q][C].
__global__ void __launch_bounds__(256) kMain(const _Float16* __restrict__ Qm,
                                             const _Float16* __restrict__ Sm,
                                             const int* __restrict__ labels,
                                             const float* __restrict__ scale_p,
                                             float* __restrict__ CM,
                                             int qBlocks, int Qtot, int S) {
  __shared__ __align__(16) float CA[32 * CASTRIDE];  // ~32.9 KB class-mass accum
  int tid = threadIdx.x;
  int bx = blockIdx.x;
  int qb = bx % qBlocks;
  int sz = bx / qBlocks;  // S-split index: 0 or 1
  int q0 = qb * 32;
  int sHalf = S >> 1;
  int sBeg = sz * sHalf;

  for (int i = tid; i < 32 * CASTRIDE; i += 256) CA[i] = 0.0f;
  __syncthreads();

  float scale = fminf(fmaxf(scale_p[0], 1.0f), 20.0f);

  int lane = tid & 31;
  int w = tid >> 5;
  int half = lane >> 4;   // lane group 0/1
  int mrow = lane & 15;   // A row / B column owned by this lane

  // A-fragment source (query f16, row-major): halves kb..kb+7 and kb+16..kb+23
  const _Float16* pa0 = Qm + (size_t)(q0 + mrow) * D_DIM + half * 8;        // q-tile 0
  const _Float16* pa1 = Qm + (size_t)(q0 + 16 + mrow) * D_DIM + half * 8;   // q-tile 1

  int npairs = sHalf >> 5;  // 32 supports per pair
  for (int p = w; p < npairs; p += 8) {
    int s0 = sBeg + p * 32;
    // B-fragment source (support f16, row-major): halves kb..kb+7 and kb+16..kb+23
    const _Float16* pb0 = Sm + (size_t)(s0 + mrow) * D_DIM + half * 8;
    const _Float16* pb1 = Sm + (size_t)(s0 + 16 + mrow) * D_DIM + half * 8;

    v8f acc00 = vzero8(), acc01 = vzero8(), acc10 = vzero8(), acc11 = vzero8();

#pragma unroll 4
    for (int k0 = 0; k0 < D_DIM; k0 += 32) {
      v8h a0lo = *(const v8h*)(pa0 + k0);
      v8h a0hi = *(const v8h*)(pa0 + k0 + 16);
      v8h a1lo = *(const v8h*)(pa1 + k0);
      v8h a1hi = *(const v8h*)(pa1 + k0 + 16);
      v8h b0lo = *(const v8h*)(pb0 + k0);
      v8h b0hi = *(const v8h*)(pb0 + k0 + 16);
      v8h b1lo = *(const v8h*)(pb1 + k0);
      v8h b1hi = *(const v8h*)(pb1 + k0 + 16);
      v16h A0 = __builtin_shufflevector(a0lo, a0hi, 0, 1, 2, 3, 4, 5, 6, 7, 8, 9, 10, 11, 12, 13, 14, 15);
      v16h A1 = __builtin_shufflevector(a1lo, a1hi, 0, 1, 2, 3, 4, 5, 6, 7, 8, 9, 10, 11, 12, 13, 14, 15);
      v16h B0 = __builtin_shufflevector(b0lo, b0hi, 0, 1, 2, 3, 4, 5, 6, 7, 8, 9, 10, 11, 12, 13, 14, 15);
      v16h B1 = __builtin_shufflevector(b1lo, b1hi, 0, 1, 2, 3, 4, 5, 6, 7, 8, 9, 10, 11, 12, 13, 14, 15);
      acc00 = __builtin_amdgcn_wmma_f32_16x16x32_f16(false, A0, false, B0, (short)0, acc00, false, false);
      acc01 = __builtin_amdgcn_wmma_f32_16x16x32_f16(false, A0, false, B1, (short)0, acc01, false, false);
      acc10 = __builtin_amdgcn_wmma_f32_16x16x32_f16(false, A1, false, B0, (short)0, acc10, false, false);
      acc11 = __builtin_amdgcn_wmma_f32_16x16x32_f16(false, A1, false, B1, (short)0, acc11, false, false);
    }

    // |sims*scale| <= 20 -> exp is safe without max subtraction.
    int lab0 = labels[s0 + mrow];
    int lab1 = labels[s0 + 16 + mrow];
#pragma unroll
    for (int i = 0; i < 8; ++i) {
      int m0 = i + half * 8;   // C/D layout: lane group selects M block
      int m1 = m0 + 16;
      atomicAdd(&CA[m0 * CASTRIDE + lab0], __expf(acc00[i] * scale));
      atomicAdd(&CA[m0 * CASTRIDE + lab1], __expf(acc01[i] * scale));
      atomicAdd(&CA[m1 * CASTRIDE + lab0], __expf(acc10[i] * scale));
      atomicAdd(&CA[m1 * CASTRIDE + lab1], __expf(acc11[i] * scale));
    }
  }
  __syncthreads();

  // Flush partials: unique (sz, q, c) per block -> plain stores, no atomics.
  for (int i = tid; i < 32 * C_CLS; i += 256) {
    int r = i >> 8;
    int c = i & 255;
    CM[((size_t)sz * Qtot + (q0 + r)) * C_CLS + c] = CA[r * CASTRIDE + c];
  }
}

// ---------------- logits + count prior + confidence softmax ----------------
__global__ void __launch_bounds__(256) kFinal(const float* __restrict__ CM,
                                              const float* __restrict__ counts,
                                              const float* __restrict__ cprior_p,
                                              float* __restrict__ out, int Qtot) {
  int tid = threadIdx.x;
  int lane = tid & 31;
  int w = tid >> 5;
  int q = blockIdx.x * 8 + w;  // one wave per query row
  float cp = cprior_p[0];

  float e[8], lg[8];
  float tot = 0.0f;
#pragma unroll
  for (int j = 0; j < 8; ++j) {
    int c = lane + 32 * j;
    e[j] = CM[(size_t)q * C_CLS + c] + CM[((size_t)Qtot + q) * C_CLS + c];
    tot += e[j];
  }
  tot = waveSum32(tot);
  float inv = 1.0f / tot;

  float mx = -3.4e38f;
#pragma unroll
  for (int j = 0; j < 8; ++j) {
    int c = lane + 32 * j;
    float l = logf(fmaxf(e[j] * inv, 1e-8f)) + cp * logf(fmaxf(counts[c], 1.0f));
    lg[j] = l;
    mx = fmaxf(mx, l);
    out[(size_t)q * C_CLS + c] = l;  // logits
  }
  mx = waveMax32(mx);
  float se = 0.0f;
#pragma unroll
  for (int j = 0; j < 8; ++j) {
    e[j] = __expf(lg[j] - mx);
    se += e[j];
  }
  se = waveSum32(se);
  float sinv = 1.0f / se;
  size_t confOff = (size_t)Qtot * C_CLS + C_CLS;
#pragma unroll
  for (int j = 0; j < 8; ++j)
    out[confOff + (size_t)q * C_CLS + lane + 32 * j] = e[j] * sinv;  // confidence

  if (blockIdx.x == 0 && tid < C_CLS)
    out[(size_t)Qtot * C_CLS + tid] = (float)tid;  // unique_classes (arange)
}

extern "C" void kernel_launch(void* const* d_in, const int* in_sizes, int n_in,
                              void* d_out, int out_size, void* d_ws, size_t ws_size,
                              hipStream_t stream) {
  const float* supF = (const float*)d_in[0];
  const int* labels = (const int*)d_in[1];
  const float* qryF = (const float*)d_in[2];
  const float* scale_p = (const float*)d_in[3];
  const float* cprior_p = (const float*)d_in[4];
  int S = in_sizes[1];            // 16384
  int Q = in_sizes[2] / D_DIM;    // 4096
  float* out = (float*)d_out;

  // Workspace layout (bytes):
  //   [0,4096)        mu_sum  f32[1024]
  //   [4096,5120)     counts  f32[256]
  //   [8192, +2*Q*C*4)   CM partials f32[2][Q][C]
  //   then support f16 [S][D], query f16 [Q][D]
  char* ws = (char*)d_ws;
  float* mu_sum = (float*)ws;
  float* counts = (float*)(ws + 4096);
  float* CM = (float*)(ws + 8192);
  size_t cmBytes = (size_t)2 * Q * C_CLS * sizeof(float);
  _Float16* sup16 = (_Float16*)(ws + 8192 + cmBytes);
  _Float16* qry16 = (_Float16*)(ws + 8192 + cmBytes + (size_t)S * D_DIM * sizeof(_Float16));

  // mu_sum + counts are contiguous floats [0..1279] in ws
  kZero<<<8, 256, 0, stream>>>(mu_sum, 1280);
  kHist<<<(S + 255) / 256, 256, 0, stream>>>(labels, counts, S);
  kMu<<<S / 64, 256, 0, stream>>>(supF, mu_sum);
  kPack<<<S, 256, 0, stream>>>(supF, sup16, mu_sum, 1.0f / (float)S);
  kPack<<<Q, 256, 0, stream>>>(qryF, qry16, mu_sum, 1.0f / (float)S);

  int qBlocks = Q / 32;  // 128
  kMain<<<qBlocks * 2, 256, 0, stream>>>(qry16, sup16, labels, scale_p, CM, qBlocks, Q, S);
  kFinal<<<Q / 8, 256, 0, stream>>>(CM, counts, cprior_p, out, Q);
}
